// RBFKernelAttention_72078141161984
// MI455X (gfx1250) — matrix operand order repaired and checked by reading, not compiled, not run on hardware
//
#include <hip/hip_runtime.h>

#define NB 4
#define LQ 2048
#define HH 8
#define EE 64
#define SS 2048
#define DD 64

typedef __attribute__((ext_vector_type(16))) _Float16 v16h;
typedef __attribute__((ext_vector_type(8)))  _Float16 v8h;
typedef __attribute__((ext_vector_type(8)))  float    v8f;

static __device__ __forceinline__ v16h ld_pair(const _Float16* p_lo, const _Float16* p_hi) {
  v8h a = *(const v8h*)p_lo;   // 16B ds_load_b128
  v8h b = *(const v8h*)p_hi;
  v16h r;
#pragma unroll
  for (int i = 0; i < 8; ++i) { r[i] = a[i]; r[8 + i] = b[i]; }
  return r;
}

static __device__ __forceinline__ v8f zero8() {
  v8f r;
#pragma unroll
  for (int i = 0; i < 8; ++i) r[i] = 0.f;
  return r;
}

__global__ __launch_bounds__(256, 2)
void rbf_attn_wmma(const float* __restrict__ q,
                   const float* __restrict__ k,
                   const float* __restrict__ v,
                   const float* __restrict__ mask,
                   float* __restrict__ out)
{
  // LDS staging (single buffered): K tile row-major f16, V tile d-major f16,
  // per-wave phi scratch for the A-operand transpose.
  __shared__ __align__(16) _Float16 kf16[32 * 64];     // 4 KB
  __shared__ __align__(16) _Float16 vt16[64 * 32];     // 4 KB (vt16[d][s])
  __shared__ __align__(16) _Float16 phis[8 * 16 * 32]; // 8 KB (per-wave 16x32)
  __shared__ float ksq_s[32];
  __shared__ float msk_s[32];
  __shared__ float qsq_s[128];

  const int n    = blockIdx.z;
  const int h    = blockIdx.y;
  const int l0   = blockIdx.x * 128;
  const int tid  = threadIdx.x;
  const int wave = tid >> 5;
  const int lane = tid & 31;
  const int m16  = lane & 15;   // N index / row-in-tile
  const int hi   = lane >> 4;   // half-wave select

  const float* qb = q + ((size_t)n * HH + h) * (size_t)LQ * EE;       // q[n][h][l][e] (flat reinterpretation)
  const float* kb = k + ((size_t)n * HH + h) * (size_t)SS * EE;       // k[n][h][s][e]
  const float* vb = v + (size_t)n * SS * HH * DD + (size_t)h * DD;    // v[n][s][h][d]
  const float* mb = mask + (size_t)n * SS;
  float* ob = out + (size_t)n * LQ * HH * DD + (size_t)h * DD;        // out[n][l][h][d]

  // ---- stage ||q||^2 for this block's 128 rows (2 threads per row) ----
  {
    const int qr  = tid >> 1;
    const int hlf = tid & 1;
    const float* p = qb + (size_t)(l0 + qr) * EE + hlf * 32;
    float s = 0.f;
#pragma unroll
    for (int i = 0; i < 32; ++i) { float x = p[i]; s += x * x; }
    s += __shfl_xor(s, 1, 32);
    if (hlf == 0) qsq_s[qr] = s;
  }
  __syncthreads();

  // ---- per-wave Q A-operands (16 rows x 64 feats -> two 16x32 f16 tiles) ----
  // A layout: elems 0..7 <-> K = 8*hi + e ; elems 8..15 <-> K = 16 + 8*hi + e
  const int qrow = l0 + wave * 16 + m16;
  const float* qr = qb + (size_t)qrow * EE;
  v16h aq0, aq1;
  {
#pragma unroll
    for (int e = 0; e < 8; ++e) aq0[e]     = (_Float16)qr[8 * hi + e];
#pragma unroll
    for (int e = 0; e < 8; ++e) aq0[8 + e] = (_Float16)qr[16 + 8 * hi + e];
#pragma unroll
    for (int e = 0; e < 8; ++e) aq1[e]     = (_Float16)qr[32 + 8 * hi + e];
#pragma unroll
    for (int e = 0; e < 8; ++e) aq1[8 + e] = (_Float16)qr[48 + 8 * hi + e];
  }
  float qs[8];
#pragma unroll
  for (int g = 0; g < 8; ++g) qs[g] = qsq_s[wave * 16 + 8 * hi + g];

  v8f acc[4];
#pragma unroll
  for (int t = 0; t < 4; ++t) acc[t] = zero8();
  float rs[8];
#pragma unroll
  for (int g = 0; g < 8; ++g) rs[g] = 0.f;

  const float tau2 = 0.0625f;  // 0.5 / sqrt(E=64)

  for (int s0 = 0; s0 < SS; s0 += 32) {
    __syncthreads();  // previous tile fully consumed
    // ---- cooperative staging: 256 threads = 32 rows x 8 chunks of 8 feats ----
    {
      const int r  = tid >> 3;
      const int c8 = (tid & 7) * 8;
      const float* kp = kb + (size_t)(s0 + r) * EE + c8;
      float x[8]; float sq = 0.f;
#pragma unroll
      for (int i = 0; i < 8; ++i) { x[i] = kp[i]; sq += x[i] * x[i]; }
      v8h hv;
#pragma unroll
      for (int i = 0; i < 8; ++i) hv[i] = (_Float16)x[i];
      *(v8h*)&kf16[r * 64 + c8] = hv;             // 16B ds_store_b128
      sq += __shfl_xor(sq, 1, 32);
      sq += __shfl_xor(sq, 2, 32);
      sq += __shfl_xor(sq, 4, 32);
      if ((tid & 7) == 0) ksq_s[r] = sq;

      const float* vp = vb + (size_t)(s0 + r) * (HH * DD) + c8;
#pragma unroll
      for (int i = 0; i < 8; ++i) vt16[(c8 + i) * 32 + r] = (_Float16)vp[i];  // transpose into d-major

      if (tid < 32) msk_s[tid] = mb[s0 + tid];

      if (s0 + 32 < SS) {  // pull next tile toward L2/L0 (global_prefetch_b8)
        __builtin_prefetch(kp + (size_t)32 * EE, 0, 1);
        __builtin_prefetch(vp + (size_t)32 * HH * DD, 0, 1);
      }
    }
    __syncthreads();

    _Float16* pw = &phis[wave * 512];  // per-wave 16x32 phi scratch

    // ---- QK^T over two 16-key subtiles; exp/clamp/mask in fp32 ----
#pragma unroll
    for (int jt = 0; jt < 2; ++jt) {
      const int j = jt * 16 + m16;           // key index within 32-tile
      const _Float16* kr = &kf16[j * 64];
      // B layout: elems 0..7 <-> K = 8*hi + e ; elems 8..15 <-> K = 16 + 8*hi + e
      v16h bk0 = ld_pair(kr + 8 * hi,      kr + 16 + 8 * hi);        // feats 0..31
      v16h bk1 = ld_pair(kr + 32 + 8 * hi, kr + 48 + 8 * hi);       // feats 32..63
      v8f c = zero8();
      c = __builtin_amdgcn_wmma_f32_16x16x32_f16(false, aq0, false, bk0, (short)0, c, false, false);
      c = __builtin_amdgcn_wmma_f32_16x16x32_f16(false, aq1, false, bk1, (short)0, c, false, false);
      const float ksqj = ksq_s[j];
      const float mj   = msk_s[j];
#pragma unroll
      for (int g = 0; g < 8; ++g) {
        float d2 = qs[g] + ksqj - 2.f * c[g];
        d2 = fmaxf(d2, 0.f);
        float ph = __expf(-tau2 * d2) * mj;
        rs[g] += ph;
        pw[(8 * hi + g) * 32 + j] = (_Float16)ph;  // transpose via per-wave LDS
      }
    }

    // ---- phi * V : A = phi (16x32), B = V^T tiles (32x16 per d-block) ----
    const _Float16* pr = &pw[m16 * 32];
    v16h ap = ld_pair(pr + 8 * hi, pr + 16 + 8 * hi);
#pragma unroll
    for (int t = 0; t < 4; ++t) {
      const _Float16* vr = &vt16[(t * 16 + m16) * 32];
      v16h bv = ld_pair(vr + 8 * hi, vr + 16 + 8 * hi);
      acc[t] = __builtin_amdgcn_wmma_f32_16x16x32_f16(false, ap, false, bv, (short)0, acc[t], false, false);
    }
  }

  // ---- rowsum reduce across the 16 N-lanes, normalize, store ----
#pragma unroll
  for (int g = 0; g < 8; ++g) {
    float r = rs[g];
    r += __shfl_xor(r, 1, 32);
    r += __shfl_xor(r, 2, 32);
    r += __shfl_xor(r, 4, 32);
    r += __shfl_xor(r, 8, 32);
    rs[g] = 1.f / (r + 1e-6f);
  }
#pragma unroll
  for (int t = 0; t < 4; ++t) {
#pragma unroll
    for (int g = 0; g < 8; ++g) {
      const int row = l0 + wave * 16 + 8 * hi + g;
      const int col = t * 16 + m16;
      ob[(size_t)row * (HH * DD) + col] = acc[t][g] * rs[g];
    }
  }
}

extern "C" void kernel_launch(void* const* d_in, const int* in_sizes, int n_in,
                              void* d_out, int out_size, void* d_ws, size_t ws_size,
                              hipStream_t stream) {
  (void)in_sizes; (void)n_in; (void)out_size; (void)d_ws; (void)ws_size;
  const float* q    = (const float*)d_in[0];
  const float* k    = (const float*)d_in[1];
  const float* v    = (const float*)d_in[2];
  const float* mask = (const float*)d_in[3];
  float* out = (float*)d_out;
  dim3 grid(LQ / 128, HH, NB);
  rbf_attn_wmma<<<grid, dim3(256), 0, stream>>>(q, k, v, mask, out);
}
